// TrackingPropagator_4818953306586
// MI455X (gfx1250) — matrix operand order repaired and checked by reading, not compiled, not run on hardware
//
#include <hip/hip_runtime.h>
#include <hip/hip_bf16.h>
#include <cstdint>

typedef _Float16 v16h __attribute__((ext_vector_type(16)));
typedef _Float16 v8h  __attribute__((ext_vector_type(8)));
typedef float    v8f  __attribute__((ext_vector_type(8)));
typedef unsigned int tdm_u32x4 __attribute__((ext_vector_type(4)));
typedef int          tdm_i32x8 __attribute__((ext_vector_type(8)));
typedef int          tdm_i32x4 __attribute__((ext_vector_type(4)));

#define WBATCH 32
#define NSLOT  300
#define NROWS  9600      // WBATCH * NSLOT
#define GRD    60
#define CELLS  3600

// ---------------------------------------------------------------- utilities
__global__ void tp_f32_to_f16(const float* __restrict__ s, _Float16* __restrict__ d, int n) {
    int i = blockIdx.x * 256 + threadIdx.x;
    if (i < n) d[i] = (_Float16)s[i];
}

// ------------------------------------------------------- conv3x3 + BN + ReLU
__global__ void tp_conv_bn_relu(const float* __restrict__ in, const float* __restrict__ w,
                                const float* __restrict__ cb, const float* __restrict__ bg,
                                const float* __restrict__ bb, float* __restrict__ out,
                                int Cin, int Cout) {
    int idx = blockIdx.x * 256 + threadIdx.x;
    int total = WBATCH * Cout * CELLS;
    if (idx >= total) return;
    int x  = idx % GRD;
    int y  = (idx / GRD) % GRD;
    int oc = (idx / CELLS) % Cout;
    int b  = idx / (CELLS * Cout);
    float acc = cb[oc];
    for (int ic = 0; ic < Cin; ++ic) {
        const float* ip = in + (size_t)(b * Cin + ic) * CELLS;
        const float* wp = w + (size_t)(oc * Cin + ic) * 9;
        #pragma unroll
        for (int ky = 0; ky < 3; ++ky) {
            int yy = y + ky - 1;
            if (yy < 0 || yy >= GRD) continue;
            #pragma unroll
            for (int kx = 0; kx < 3; ++kx) {
                int xx = x + kx - 1;
                if (xx < 0 || xx >= GRD) continue;
                acc += ip[yy * GRD + xx] * wp[ky * 3 + kx];
            }
        }
    }
    float s = bg[oc] * rsqrtf(1.0f + 1e-5f);
    out[idx] = fmaxf(acc * s + bb[oc], 0.0f);
}

// ----------------------------------------------------------- global avg pool
__global__ void tp_pool(const float* __restrict__ gf, float* __restrict__ gfeat) {
    int w = threadIdx.x >> 5, lane = threadIdx.x & 31;
    int gi = blockIdx.x * 8 + w;               // b*64 + ch, 2048 total
    const float* p = gf + (size_t)gi * CELLS;
    float s = 0.0f;
    for (int i = lane; i < CELLS; i += 32) s += p[i];
    #pragma unroll
    for (int off = 16; off; off >>= 1) s += __shfl_xor(s, off);
    if (lane == 0) gfeat[gi] = s * (1.0f / CELLS);
}

// ---------------------- per-box stats + motion MLP + box MLP + id embedding
// One wave per (b,n) box; writes f16 row (n*32+b) of A_obj [9600,160]
__global__ void tp_perbox(const float* __restrict__ mv, const float* __restrict__ boxes,
                          const int* __restrict__ trk,
                          const float* __restrict__ aw1, const float* __restrict__ ab1,
                          const float* __restrict__ aw2, const float* __restrict__ ab2,
                          const float* __restrict__ bw1, const float* __restrict__ bb1,
                          const float* __restrict__ bw2, const float* __restrict__ bb2,
                          const float* __restrict__ idtab, _Float16* __restrict__ Aobj) {
    int w = threadIdx.x >> 5, lane = threadIdx.x & 31;
    int box = blockIdx.x * 8 + w;              // 9600 = 1200*8 exactly
    int b = box / NSLOT, n = box % NSLOT;
    const float* bx = boxes + ((size_t)b * NSLOT + n) * 4;
    float b0 = bx[0], b1 = bx[1], b2 = bx[2], b3 = bx[3];
    const float HI = 60.0f - 1e-6f;
    float g0 = fminf(fmaxf(b0 * 60.0f, 0.0f), HI);
    float g1 = fminf(fmaxf(b1 * 60.0f, 0.0f), HI);
    float g2 = fminf(fmaxf(b2 * 60.0f, 0.0f), HI);
    float g3 = fminf(fmaxf(b3 * 60.0f, 0.0f), HI);
    float x1f = floorf(g0), y1f = floorf(g1);
    float x2f = fmaxf(x1f + 1.0f, ceilf(g2));
    float y2f = fmaxf(y1f + 1.0f, ceilf(g3));
    int x1 = (int)x1f, y1 = (int)y1f, x2 = (int)x2f, y2 = (int)y2f;
    int wx = x2 - x1, wy = y2 - y1;
    int cnt = wx * wy;
    float s1[2] = {0, 0}, s2[2] = {0, 0}, mn[2] = {1e9f, 1e9f}, mx[2] = {-1e9f, -1e9f};
    for (int idx = lane; idx < cnt; idx += 32) {
        int yy = y1 + idx / wx, xx = x1 + idx % wx;
        #pragma unroll
        for (int c = 0; c < 2; ++c) {
            float v = mv[((size_t)(b * 2 + c) * GRD + yy) * GRD + xx];
            s1[c] += v; s2[c] += v * v;
            mn[c] = fminf(mn[c], v); mx[c] = fmaxf(mx[c], v);
        }
    }
    #pragma unroll
    for (int off = 16; off; off >>= 1) {
        #pragma unroll
        for (int c = 0; c < 2; ++c) {
            s1[c] += __shfl_xor(s1[c], off);
            s2[c] += __shfl_xor(s2[c], off);
            mn[c] = fminf(mn[c], __shfl_xor(mn[c], off));
            mx[c] = fmaxf(mx[c], __shfl_xor(mx[c], off));
        }
    }
    float cf = (float)cnt;
    float stats[8];
    #pragma unroll
    for (int c = 0; c < 2; ++c) {
        float mean = s1[c] / cf;
        float var  = (s2[c] - cf * mean * mean) / fmaxf(cf - 1.0f, 1.0f);
        float sd   = (cnt > 1) ? sqrtf(fmaxf(var, 1e-12f)) : 0.0f;
        stats[c] = mean; stats[2 + c] = sd;
        stats[4 + 2 * c] = mn[c]; stats[5 + 2 * c] = mx[c];
    }
    size_t rowBase = ((size_t)n * WBATCH + b) * 160;
    // motion MLP 8->32->64  (hidden unit = lane)
    float hs = ab1[lane];
    #pragma unroll
    for (int k = 0; k < 8; ++k) hs += aw1[lane * 8 + k] * stats[k];
    float hval = fmaxf(hs, 0.0f);
    #pragma unroll
    for (int oo = 0; oo < 2; ++oo) {
        int o = lane + oo * 32;
        float acc = ab2[o];
        for (int k = 0; k < 32; ++k) acc += aw2[o * 32 + k] * __shfl(hval, k);
        Aobj[rowBase + 96 + o] = (_Float16)acc;
    }
    // box MLP 4->32->64
    float bh = bb1[lane];
    bh += bw1[lane * 4 + 0] * b0 + bw1[lane * 4 + 1] * b1 +
          bw1[lane * 4 + 2] * b2 + bw1[lane * 4 + 3] * b3;
    bh = fmaxf(bh, 0.0f);
    #pragma unroll
    for (int oo = 0; oo < 2; ++oo) {
        int o = lane + oo * 32;
        float acc = bb2[o];
        for (int k = 0; k < 32; ++k) acc += bw2[o * 32 + k] * __shfl(bh, k);
        Aobj[rowBase + o] = (_Float16)acc;
    }
    // id embedding (32 wide, one per lane)
    int tv = trk[b * NSLOT + n];
    tv = tv < 0 ? 0 : (tv > 999 ? 999 : tv);
    Aobj[rowBase + 64 + lane] = (_Float16)idtab[(size_t)tv * 32 + lane];
}

// -------------------------------- fill global-feature cols 128..191 of X0
__global__ void tp_gfill(const float* __restrict__ gfeat, _Float16* __restrict__ X0) {
    int idx = blockIdx.x * 256 + threadIdx.x;   // 9600*64 exact
    int row = idx >> 6, c = idx & 63;
    X0[(size_t)row * 192 + 128 + c] = (_Float16)gfeat[(row & 31) * 64 + c];
}

// ------------------------------------------------------- WMMA fragment utils
__device__ inline v16h tp_mk_a(const _Float16* p) {   // p -> A + row*ld + k0 + hi*8
    v8h lo = *reinterpret_cast<const v8h*>(p);
    v8h h8 = *reinterpret_cast<const v8h*>(p + 16);
    v16h a;
    #pragma unroll
    for (int i = 0; i < 8; ++i) { a[i] = lo[i]; a[8 + i] = h8[i]; }
    return a;
}

// ------------------------------------------- WMMA GEMM (4 waves, 16x64 tile)
// C[M,N] = act(A[M,K] @ W[N,K]^T + bias)   A,W f16 row-major, K%32==0, K<=256
// The 16-row A tile (contiguous 16*K f16) is staged into LDS once per block
// by the Tensor Data Mover (wave 0 issues tensor_load_to_lds + waits
// TENSORcnt, barrier releases the other 3 waves); all 4 waves reuse it.
// SMODE 0: f32 out with ldc    SMODE 1: f16 out with ldc
// SMODE 2: f32 out mapped  out[((row&31)*NSLOT + row>>5)*N + col]
template<int ACT, int SMODE>
__global__ __launch_bounds__(128)
void tp_gemm16(const _Float16* __restrict__ A, const _Float16* __restrict__ W,
               const float* __restrict__ bias, float* __restrict__ outF,
               _Float16* __restrict__ outH, int M, int N, int K, int ldc) {
    __shared__ __align__(32) _Float16 Atile[16 * 256];
    const int tid = threadIdx.x;
    const int w = tid >> 5, lane = tid & 31;
    const int r = lane & 15, hi = lane >> 4;
    const int m0 = blockIdx.y * 16;
    const int n0 = blockIdx.x * 64 + w * 16;
    const int col = n0 + r;

#if defined(__gfx1250__) && __has_builtin(__builtin_amdgcn_tensor_load_to_lds)
    if (w == 0) {
        unsigned long long ga = (unsigned long long)(uintptr_t)(A + (size_t)m0 * K);
        unsigned int lds_off  = (unsigned int)(uintptr_t)(&Atile[0]);
        unsigned int nelem    = (unsigned int)(16 * K);   // <= 4096 elements
        tdm_u32x4 g0;
        g0[0] = 1u;                                               // count=1, user mode
        g0[1] = lds_off;                                          // lds_addr (bytes)
        g0[2] = (unsigned int)(ga & 0xffffffffull);               // global_addr[31:0]
        g0[3] = (unsigned int)((ga >> 32) & 0x01ffffffull)        // global_addr[56:32]
              | (2u << 30);                                       // type = 2 (image)
        tdm_i32x8 g1;
        g1[0] = (int)(1u << 16);                  // data_size=1 -> 2-byte elements
        g1[1] = (int)((nelem & 0xffffu) << 16);   // tensor_dim0[15:0]
        g1[2] = (int)((nelem >> 16) | (1u << 16));// tensor_dim0[31:16] | tensor_dim1=1
        g1[3] = (int)((nelem & 0xffffu) << 16);   // tile_dim0 = 16*K (1-D tile)
        g1[4] = 0;                                // tile_dim1=0 (unused), tile_dim2=0
        g1[5] = (int)nelem;                       // tensor_dim0_stride[31:0]
        g1[6] = 0;
        g1[7] = 0;
        tdm_i32x4 gz;
        gz[0] = 0; gz[1] = 0; gz[2] = 0; gz[3] = 0;
        tdm_i32x8 gz8;
        gz8[0] = 0; gz8[1] = 0; gz8[2] = 0; gz8[3] = 0;
        gz8[4] = 0; gz8[5] = 0; gz8[6] = 0; gz8[7] = 0;
        __builtin_amdgcn_tensor_load_to_lds(g0, g1, gz, gz, gz8, 0);
        __builtin_amdgcn_s_wait_tensorcnt(0);
    }
    __syncthreads();
#else
    for (int i = tid; i < 16 * K; i += 128) Atile[i] = A[(size_t)m0 * K + i];
    __syncthreads();
#endif

    v8f acc;
    #pragma unroll
    for (int i = 0; i < 8; ++i) acc[i] = 0.0f;
    const _Float16* wrow = W + (size_t)(col < N ? col : N - 1) * K;
    for (int k0 = 0; k0 < K; k0 += 32) {
        if (k0 + 32 < K) __builtin_prefetch(wrow + k0 + 32, 0, 0);
        v16h a = tp_mk_a(Atile + r * K + k0 + hi * 8);
        v16h b = tp_mk_a(wrow + k0 + hi * 8);
        if (col >= N) {
            #pragma unroll
            for (int i = 0; i < 16; ++i) b[i] = (_Float16)0;
        }
        acc = __builtin_amdgcn_wmma_f32_16x16x32_f16(false, a, false, b, (short)0,
                                                     acc, false, false);
    }
    float bv = (col < N) ? bias[col] : 0.0f;
    #pragma unroll
    for (int j = 0; j < 8; ++j) {
        int row = m0 + hi * 8 + j;
        float v = acc[j] + bv;
        if (ACT == 1) v = fmaxf(v, 0.0f);
        if (ACT == 2) v = 1.0f / (1.0f + __expf(-v));
        if (col < N) {
            if (SMODE == 0)      outF[(size_t)row * ldc + col] = v;
            else if (SMODE == 1) outH[(size_t)row * ldc + col] = (_Float16)v;
            else {
                int bb = row & 31, tt = row >> 5;
                outF[((size_t)bb * NSLOT + tt) * N + col] = v;
            }
        }
    }
}

// ------------------------------------------- persistent 1-WG LSTM layer step
// pre: [T*32, 512] f32 (x@wih^T + b, rows t*32+batch); whh f16 [512,128]
// Y: f16 [T*32, 128];  hOut/cOut: f32 [32,128] (final states)
__global__ __launch_bounds__(256)
void tp_lstm_layer(const float* __restrict__ pre, const _Float16* __restrict__ whh,
                   _Float16* __restrict__ Y, float* __restrict__ hOut,
                   float* __restrict__ cOut, int T) {
    __shared__ __align__(32) _Float16 hsh[32 * 128];
    const int tid = threadIdx.x;
    const int w = tid >> 5, lane = tid & 31;
    const int r = lane & 15, hi = lane >> 4;
    // whh fragments live in registers for all 300 steps (4 gates x 4 K-tiles)
    v16h bf[4][4];
    #pragma unroll
    for (int g = 0; g < 4; ++g)
        #pragma unroll
        for (int kt = 0; kt < 4; ++kt)
            bf[g][kt] = tp_mk_a(
                whh + (size_t)(g * 128 + w * 16 + r) * 128 + kt * 32 + hi * 8);
    for (int i = tid; i < 32 * 128; i += 256) hsh[i] = (_Float16)0;
    float cs[16];
    #pragma unroll
    for (int i = 0; i < 16; ++i) cs[i] = 0.0f;
    __syncthreads();
    const int colIdx = w * 16 + r;      // hidden unit owned by this lane
    for (int t = 0; t < T; ++t) {
        v16h af[2][4];
        #pragma unroll
        for (int mt = 0; mt < 2; ++mt)
            #pragma unroll
            for (int kt = 0; kt < 4; ++kt)
                af[mt][kt] = tp_mk_a(hsh + (mt * 16 + r) * 128 + kt * 32 + hi * 8);
        __syncthreads();     // everyone read h before it gets overwritten
        #pragma unroll
        for (int mt = 0; mt < 2; ++mt) {
            v8f za[4];
            #pragma unroll
            for (int g = 0; g < 4; ++g) {
                #pragma unroll
                for (int j = 0; j < 8; ++j) {
                    int bb = mt * 16 + hi * 8 + j;
                    za[g][j] = pre[((size_t)t * 32 + bb) * 512 + g * 128 + colIdx];
                }
                #pragma unroll
                for (int kt = 0; kt < 4; ++kt)
                    za[g] = __builtin_amdgcn_wmma_f32_16x16x32_f16(
                        false, af[mt][kt], false, bf[g][kt], (short)0, za[g], false, false);
            }
            #pragma unroll
            for (int j = 0; j < 8; ++j) {
                int bb = mt * 16 + hi * 8 + j;
                float iv = 1.0f / (1.0f + __expf(-za[0][j]));
                float fv = 1.0f / (1.0f + __expf(-za[1][j]));
                float gv = tanhf(za[2][j]);
                float ov = 1.0f / (1.0f + __expf(-za[3][j]));
                float c_ = fv * cs[mt * 8 + j] + iv * gv;
                cs[mt * 8 + j] = c_;
                float h_ = ov * tanhf(c_);
                _Float16 h16 = (_Float16)h_;
                hsh[bb * 128 + colIdx] = h16;
                Y[((size_t)t * 32 + bb) * 128 + colIdx] = h16;
                if (t == T - 1) {
                    hOut[bb * 128 + colIdx] = h_;
                    cOut[bb * 128 + colIdx] = c_;
                }
            }
        }
        __syncthreads();     // h writes visible before next step's reads
    }
}

// ---------------------------------------------------------------------------
extern "C" void kernel_launch(void* const* d_in, const int* in_sizes, int n_in,
                              void* d_out, int out_size, void* d_ws, size_t ws_size,
                              hipStream_t stream) {
    (void)in_sizes; (void)n_in; (void)out_size; (void)ws_size;
    const float* mv      = (const float*)d_in[0];
    const float* boxes   = (const float*)d_in[1];
    const int*   trk     = (const int*)d_in[2];
    const float* c1w = (const float*)d_in[3],  *c1b = (const float*)d_in[4];
    const float* b1g = (const float*)d_in[5],  *b1b = (const float*)d_in[6];
    const float* c2w = (const float*)d_in[7],  *c2b = (const float*)d_in[8];
    const float* b2g = (const float*)d_in[9],  *b2b = (const float*)d_in[10];
    const float* c3w = (const float*)d_in[11], *c3b = (const float*)d_in[12];
    const float* b3g = (const float*)d_in[13], *b3b = (const float*)d_in[14];
    const float* aw1 = (const float*)d_in[15], *ab1 = (const float*)d_in[16];
    const float* aw2 = (const float*)d_in[17], *ab2 = (const float*)d_in[18];
    const float* bw1 = (const float*)d_in[19], *bb1 = (const float*)d_in[20];
    const float* bw2 = (const float*)d_in[21], *bb2 = (const float*)d_in[22];
    const float* idtab = (const float*)d_in[23];
    const float* projw = (const float*)d_in[24], *projb = (const float*)d_in[25];
    const float* l0wih = (const float*)d_in[27], *l0whh = (const float*)d_in[28];
    const float* l0b   = (const float*)d_in[29];
    const float* l1wih = (const float*)d_in[30], *l1whh = (const float*)d_in[31];
    const float* l1b   = (const float*)d_in[32];
    const float* hbw1 = (const float*)d_in[33], *hbb1 = (const float*)d_in[34];
    const float* hbw2 = (const float*)d_in[35], *hbb2 = (const float*)d_in[36];
    const float* hcw1 = (const float*)d_in[37], *hcb1 = (const float*)d_in[38];
    const float* hcw2 = (const float*)d_in[39], *hcb2 = (const float*)d_in[40];
    const float* htw1 = (const float*)d_in[41], *htb1 = (const float*)d_in[42];
    const float* htw2 = (const float*)d_in[43], *htb2 = (const float*)d_in[44];
    float* out = (float*)d_out;

    char* ws = (char*)d_ws;
    size_t off = 0;
    auto alloc = [&](size_t bytes) -> char* {
        char* p = ws + off;
        off += (bytes + 255) & ~(size_t)255;
        return p;
    };
    float* X1   = (float*)alloc((size_t)WBATCH * 32 * CELLS * 4);  // conv1 out, later head hidden (f16)
    float* X2   = (float*)alloc((size_t)WBATCH * 64 * CELLS * 4);  // conv2 out, later pre1
    float* GF   = (float*)alloc((size_t)WBATCH * 64 * CELLS * 4);  // conv3 out, later pre0
    float* gfeat = (float*)alloc(WBATCH * 64 * 4);
    _Float16* Aobj = (_Float16*)alloc((size_t)NROWS * 160 * 2);
    _Float16* X0   = (_Float16*)alloc((size_t)NROWS * 192 * 2);
    _Float16* Y0   = (_Float16*)alloc((size_t)NROWS * 128 * 2);
    _Float16* Y1   = (_Float16*)alloc((size_t)NROWS * 128 * 2);
    _Float16* w_proj  = (_Float16*)alloc(128 * 160 * 2);
    _Float16* w_l0wih = (_Float16*)alloc(512 * 192 * 2);
    _Float16* w_l0whh = (_Float16*)alloc(512 * 128 * 2);
    _Float16* w_l1wih = (_Float16*)alloc(512 * 128 * 2);
    _Float16* w_l1whh = (_Float16*)alloc(512 * 128 * 2);
    _Float16* w_hb1 = (_Float16*)alloc(256 * 128 * 2);
    _Float16* w_hb2 = (_Float16*)alloc(4 * 256 * 2);
    _Float16* w_hc1 = (_Float16*)alloc(256 * 128 * 2);
    _Float16* w_hc2 = (_Float16*)alloc(2 * 256 * 2);
    _Float16* w_ht1 = (_Float16*)alloc(256 * 128 * 2);
    _Float16* w_ht2 = (_Float16*)alloc(1000 * 256 * 2);
    float* pre0 = GF;                 // GF consumed by pool before pre0 written
    float* pre1 = X2;                 // X2 dead after conv3
    _Float16* Hh = (_Float16*)X1;     // X1 dead after conv2

    auto c16 = [&](const float* s, _Float16* d, int n) {
        tp_f32_to_f16<<<(n + 255) / 256, 256, 0, stream>>>(s, d, n);
    };
    c16(projw, w_proj, 128 * 160);
    c16(l0wih, w_l0wih, 512 * 192);  c16(l0whh, w_l0whh, 512 * 128);
    c16(l1wih, w_l1wih, 512 * 128);  c16(l1whh, w_l1whh, 512 * 128);
    c16(hbw1, w_hb1, 256 * 128);     c16(hbw2, w_hb2, 4 * 256);
    c16(hcw1, w_hc1, 256 * 128);     c16(hcw2, w_hc2, 2 * 256);
    c16(htw1, w_ht1, 256 * 128);     c16(htw2, w_ht2, 1000 * 256);

    // grid encoder
    tp_conv_bn_relu<<<(WBATCH * 32 * CELLS + 255) / 256, 256, 0, stream>>>(mv, c1w, c1b, b1g, b1b, X1, 2, 32);
    tp_conv_bn_relu<<<(WBATCH * 64 * CELLS + 255) / 256, 256, 0, stream>>>(X1, c2w, c2b, b2g, b2b, X2, 32, 64);
    tp_conv_bn_relu<<<(WBATCH * 64 * CELLS + 255) / 256, 256, 0, stream>>>(X2, c3w, c3b, b3g, b3b, GF, 64, 64);
    tp_pool<<<256, 256, 0, stream>>>(GF, gfeat);

    // per-box stats + small MLPs + id embedding -> Aobj [9600,160]
    tp_perbox<<<1200, 256, 0, stream>>>(mv, boxes, trk, aw1, ab1, aw2, ab2,
                                        bw1, bb1, bw2, bb2, idtab, Aobj);

    // slots = Aobj @ proj_w^T + proj_b -> X0 cols 0..127 ; cols 128..191 = gfeat
    tp_gemm16<0, 1><<<dim3(2, 600), 128, 0, stream>>>(Aobj, w_proj, projb, nullptr, X0, NROWS, 128, 160, 192);
    tp_gfill<<<NROWS * 64 / 256, 256, 0, stream>>>(gfeat, X0);

    // LSTM layer 0
    tp_gemm16<0, 0><<<dim3(8, 600), 128, 0, stream>>>(X0, w_l0wih, l0b, pre0, nullptr, NROWS, 512, 192, 512);
    tp_lstm_layer<<<1, 256, 0, stream>>>(pre0, w_l0whh, Y0, out + 9657600, out + 9665792, NSLOT);
    // LSTM layer 1
    tp_gemm16<0, 0><<<dim3(8, 600), 128, 0, stream>>>(Y0, w_l1wih, l1b, pre1, nullptr, NROWS, 512, 128, 512);
    tp_lstm_layer<<<1, 256, 0, stream>>>(pre1, w_l1whh, Y1, out + 9657600 + 4096, out + 9665792 + 4096, NSLOT);

    // heads: box (sigmoid), cls, tid -- hidden buffer reused sequentially
    tp_gemm16<1, 1><<<dim3(4, 600), 128, 0, stream>>>(Y1, w_hb1, hbb1, nullptr, Hh, NROWS, 256, 128, 256);
    tp_gemm16<2, 2><<<dim3(1, 600), 128, 0, stream>>>(Hh, w_hb2, hbb2, out + 0, nullptr, NROWS, 4, 256, 0);
    tp_gemm16<1, 1><<<dim3(4, 600), 128, 0, stream>>>(Y1, w_hc1, hcb1, nullptr, Hh, NROWS, 256, 128, 256);
    tp_gemm16<0, 2><<<dim3(1, 600), 128, 0, stream>>>(Hh, w_hc2, hcb2, out + 38400, nullptr, NROWS, 2, 256, 0);
    tp_gemm16<1, 1><<<dim3(4, 600), 128, 0, stream>>>(Y1, w_ht1, htb1, nullptr, Hh, NROWS, 256, 128, 256);
    tp_gemm16<0, 2><<<dim3(16, 600), 128, 0, stream>>>(Hh, w_ht2, htb2, out + 57600, nullptr, NROWS, 1000, 256, 0);
}
